// GlobalPointerHeadZeroShot_34643206209683
// MI455X (gfx1250) — matrix operand order repaired and checked by reading, not compiled, not run on hardware
//
#include <hip/hip_runtime.h>
#include <hip/hip_bf16.h>
#include <stdint.h>

// ---------------------------------------------------------------------------
// GlobalPointerHeadZeroShot for MI455X (gfx1250, wave32, WMMA bf16)
//   out[2, 8, 2048, 2048] = stack(gp, gp_cls)
// Pipeline:
//   1) fp32 -> bf16 conversions (feature, 4 weight matrices)
//   2) projection GEMM (bf16 WMMA, f32 accum) + bias + RoPE -> bf16 Q/K in ws
//   3) logits GEMM Q@K^T (bf16 WMMA, f32 accum) + mask + scale -> f32 out
//      128x256 block tile, 64x64 wave tile: 1.0 ds_load_b128 per WMMA.
// ---------------------------------------------------------------------------

typedef __attribute__((ext_vector_type(16))) __bf16 v16bf;
typedef __attribute__((ext_vector_type(8)))  float  v8f;

#define CDIM 768
#define NTOK 2048
#define BSZ  8
#define MTOT (BSZ * NTOK)
#define BK   32
#define LDSS 40            // LDS row stride in ushorts (80 B, 16B-aligned, padded)

#define SCALE_F   0.03608439182435161f   /* 1/sqrt(768) */
#define MASKPEN_F 3.6084391824351614e8f  /* 1e10/sqrt(768) */

__device__ __forceinline__ unsigned short f2bf(float f) {
    unsigned int u = __float_as_uint(f);
    u += 0x7FFFu + ((u >> 16) & 1u);      // round to nearest even
    return (unsigned short)(u >> 16);
}

union FragAB { v16bf v; uint4 q[2]; };

// ------------------------- conversion kernels ------------------------------

__global__ __launch_bounds__(256) void cvt_feature_kernel(
    const float* __restrict__ x, unsigned short* __restrict__ y) {
    size_t i = ((size_t)blockIdx.x * blockDim.x + threadIdx.x) * 4;
    float4 v = *(const float4*)(x + i);
    uint2 p;
    p.x = (unsigned)f2bf(v.x) | ((unsigned)f2bf(v.y) << 16);
    p.y = (unsigned)f2bf(v.z) | ((unsigned)f2bf(v.w) << 16);
    *(uint2*)(y + i) = p;
}

__global__ __launch_bounds__(256) void cvt_weights_kernel(
    const float* __restrict__ w0, const float* __restrict__ w1,
    const float* __restrict__ w2, const float* __restrict__ w3,
    unsigned short* __restrict__ out) {
    const float* w = (blockIdx.y == 0) ? w0 : (blockIdx.y == 1) ? w1
                   : (blockIdx.y == 2) ? w2 : w3;
    size_t i = ((size_t)blockIdx.x * blockDim.x + threadIdx.x) * 4;
    float4 v = *(const float4*)(w + i);
    uint2 p;
    p.x = (unsigned)f2bf(v.x) | ((unsigned)f2bf(v.y) << 16);
    p.y = (unsigned)f2bf(v.z) | ((unsigned)f2bf(v.w) << 16);
    *(uint2*)(out + (size_t)blockIdx.y * CDIM * CDIM + i) = p;
}

// ------------------------- projection + RoPE -------------------------------
// grid: (CDIM/128, MTOT/128, 4)   z: 0=Q, 1=K, 2=Q_cls, 3=K_cls
// block: 256 threads = 8 wave32; wave grid 4(M) x 2(N); wave tile 32x64.

__global__ __launch_bounds__(256) void proj_rope_kernel(
    const unsigned short* __restrict__ X,     // [MTOT][C] bf16
    const unsigned short* __restrict__ Wall,  // 4 x [C][C] bf16
    const float* __restrict__ b0, const float* __restrict__ b1,
    const float* __restrict__ b2, const float* __restrict__ b3,
    unsigned short* __restrict__ QK)          // 4 x [MTOT][C] bf16
{
    __shared__ alignas(16) unsigned short As[128 * LDSS];
    __shared__ alignas(16) unsigned short Bs[128 * LDSS];

    const int z = blockIdx.z;
    const unsigned short* W = Wall + (size_t)z * CDIM * CDIM;
    const float* bias = (z == 0) ? b0 : (z == 1) ? b1 : (z == 2) ? b2 : b3;
    unsigned short* out = QK + (size_t)z * MTOT * CDIM;

    const int m0 = blockIdx.y * 128;
    const int n0 = blockIdx.x * 128;
    const int tid   = threadIdx.x;
    const int lane  = tid & 31;
    const int wv    = tid >> 5;
    const int waveM = (wv & 3) * 32;
    const int waveN = (wv >> 2) * 64;
    const int lhalf = (lane >> 4) & 1;   // 0: lanes 0-15, 1: lanes 16-31
    const int lr    = lane & 15;

    v8f acc[2][4];
    const v8f vzero = {0.f, 0.f, 0.f, 0.f, 0.f, 0.f, 0.f, 0.f};
    #pragma unroll
    for (int i = 0; i < 2; ++i)
        #pragma unroll
        for (int j = 0; j < 4; ++j) acc[i][j] = vzero;

    for (int k0 = 0; k0 < CDIM; k0 += BK) {
        __syncthreads();
        #pragma unroll
        for (int i = 0; i < 2; ++i) {
            int item = tid + i * 256;          // 512 uint4 per tile
            int row  = item >> 2;
            int c4   = (item & 3) * 8;         // ushort offset, 16B granules
            *(uint4*)&As[row * LDSS + c4] =
                *(const uint4*)(X + (size_t)(m0 + row) * CDIM + k0 + c4);
            *(uint4*)&Bs[row * LDSS + c4] =
                *(const uint4*)(W + (size_t)(n0 + row) * CDIM + k0 + c4);
        }
        __syncthreads();

        FragAB a[2], b[4];
        #pragma unroll
        for (int i = 0; i < 2; ++i) {
            int r = waveM + i * 16 + lr;
            a[i].q[0] = *(const uint4*)&As[r * LDSS + (lhalf ? 8 : 0)];
            a[i].q[1] = *(const uint4*)&As[r * LDSS + (lhalf ? 24 : 16)];
        }
        #pragma unroll
        for (int j = 0; j < 4; ++j) {
            int r = waveN + j * 16 + lr;
            int base = lhalf ? 8 : 0;
            b[j].q[0] = *(const uint4*)&Bs[r * LDSS + base];
            b[j].q[1] = *(const uint4*)&Bs[r * LDSS + base + 16];
        }
        #pragma unroll
        for (int i = 0; i < 2; ++i)
            #pragma unroll
            for (int j = 0; j < 4; ++j)
                acc[i][j] = __builtin_amdgcn_wmma_f32_16x16x32_bf16(
                    false, a[i].v, false, b[j].v, (short)0, acc[i][j],
                    false, false);
    }

    // Epilogue: bias + RoPE (interleaved sin/cos), write bf16.
    #pragma unroll
    for (int i = 0; i < 2; ++i) {
        #pragma unroll
        for (int j = 0; j < 4; ++j) {
            int d = n0 + waveN + j * 16 + lr;          // channel index
            float bv  = bias[d];
            int   hlf = d >> 1;
            float inv = __expf(-0.02398526139f * (float)hlf); // 10000^(-2h/C)
            float sgn = (d & 1) ? 1.0f : -1.0f;        // x2 = +/- partner
            #pragma unroll
            for (int g = 0; g < 8; ++g) {
                int   m = m0 + waveM + i * 16 + g + lhalf * 8;
                float q = acc[i][j][g] + bv;
                float partner = __shfl_xor(q, 1, 32);  // channel d^1 value
                float pos = (float)(m & (NTOK - 1));
                float s, c;
                __sincosf(pos * inv, &s, &c);
                float val = q * c + sgn * partner * s;
                out[(size_t)m * CDIM + d] = f2bf(val);
            }
        }
    }
}

// ------------------------- logits Q @ K^T ----------------------------------
// grid: (NTOK/256, NTOK/128, 16)   z = head*8 + batch
// block: 256 threads = 8 wave32; wave grid 2(M) x 4(N); wave tile 64x64.
// Per wave per k-step: 16 ds_load_b128 feed 16 v_wmma (1.0 DS/WMMA).

__global__ __launch_bounds__(256) void logits_kernel(
    const unsigned short* __restrict__ QK,  // 4 x [MTOT][C] bf16
    float* __restrict__ out)                // [2, 8, NTOK, NTOK] f32
{
    __shared__ alignas(16) unsigned short As[128 * LDSS];
    __shared__ alignas(16) unsigned short Bs[256 * LDSS];

    const int z = blockIdx.z;
    const int head = z >> 3;
    const int bat  = z & 7;
    const unsigned short* Q =
        QK + ((size_t)(head ? 2 : 0) * MTOT + (size_t)bat * NTOK) * CDIM;
    const unsigned short* K =
        QK + ((size_t)(head ? 3 : 1) * MTOT + (size_t)bat * NTOK) * CDIM;
    float* O = out + (size_t)z * NTOK * NTOK;

    const int m0 = blockIdx.y * 128;
    const int n0 = blockIdx.x * 256;
    const int tid   = threadIdx.x;
    const int lane  = tid & 31;
    const int wv    = tid >> 5;
    const int waveM = (wv & 1) * 64;     // 2 waves in M
    const int waveN = (wv >> 1) * 64;    // 4 waves in N
    const int lhalf = (lane >> 4) & 1;
    const int lr    = lane & 15;

    v8f acc[4][4];
    const v8f vzero = {0.f, 0.f, 0.f, 0.f, 0.f, 0.f, 0.f, 0.f};
    #pragma unroll
    for (int i = 0; i < 4; ++i)
        #pragma unroll
        for (int j = 0; j < 4; ++j) acc[i][j] = vzero;

    for (int k0 = 0; k0 < CDIM; k0 += BK) {
        __syncthreads();
        // A tile: 128 rows x 32 bf16 = 512 uint4
        #pragma unroll
        for (int i = 0; i < 2; ++i) {
            int item = tid + i * 256;
            int row  = item >> 2;
            int c4   = (item & 3) * 8;
            *(uint4*)&As[row * LDSS + c4] =
                *(const uint4*)(Q + (size_t)(m0 + row) * CDIM + k0 + c4);
        }
        // B tile: 256 rows x 32 bf16 = 1024 uint4
        #pragma unroll
        for (int i = 0; i < 4; ++i) {
            int item = tid + i * 256;
            int row  = item >> 2;
            int c4   = (item & 3) * 8;
            *(uint4*)&Bs[row * LDSS + c4] =
                *(const uint4*)(K + (size_t)(n0 + row) * CDIM + k0 + c4);
        }
        __syncthreads();

        FragAB a[4], b[4];
        #pragma unroll
        for (int i = 0; i < 4; ++i) {
            int r = waveM + i * 16 + lr;
            a[i].q[0] = *(const uint4*)&As[r * LDSS + (lhalf ? 8 : 0)];
            a[i].q[1] = *(const uint4*)&As[r * LDSS + (lhalf ? 24 : 16)];
        }
        #pragma unroll
        for (int j = 0; j < 4; ++j) {
            int r = waveN + j * 16 + lr;
            int base = lhalf ? 8 : 0;
            b[j].q[0] = *(const uint4*)&Bs[r * LDSS + base];
            b[j].q[1] = *(const uint4*)&Bs[r * LDSS + base + 16];
        }
        #pragma unroll
        for (int i = 0; i < 4; ++i)
            #pragma unroll
            for (int j = 0; j < 4; ++j)
                acc[i][j] = __builtin_amdgcn_wmma_f32_16x16x32_bf16(
                    false, a[i].v, false, b[j].v, (short)0, acc[i][j],
                    false, false);
    }

    // Epilogue: (logits - mask*1e10) / sqrt(C)
    #pragma unroll
    for (int i = 0; i < 4; ++i) {
        #pragma unroll
        for (int j = 0; j < 4; ++j) {
            int n = n0 + waveN + j * 16 + lr;
            #pragma unroll
            for (int g = 0; g < 8; ++g) {
                int m = m0 + waveM + i * 16 + g + lhalf * 8;
                float v = acc[i][j][g] * SCALE_F - ((n < m) ? MASKPEN_F : 0.0f);
                O[(size_t)m * NTOK + n] = v;
            }
        }
    }
}

// ------------------------------- launcher ----------------------------------

extern "C" void kernel_launch(void* const* d_in, const int* in_sizes, int n_in,
                              void* d_out, int out_size, void* d_ws, size_t ws_size,
                              hipStream_t stream) {
    const float* feat = (const float*)d_in[0];
    const float* Wq   = (const float*)d_in[1];
    const float* bq   = (const float*)d_in[2];
    const float* Wk   = (const float*)d_in[3];
    const float* bk   = (const float*)d_in[4];
    const float* Wqc  = (const float*)d_in[5];
    const float* bqc  = (const float*)d_in[6];
    const float* Wkc  = (const float*)d_in[7];
    const float* bkc  = (const float*)d_in[8];

    unsigned short* X16 = (unsigned short*)d_ws;                    // 25.2 MB
    unsigned short* W16 = X16 + (size_t)MTOT * CDIM;                // 4.7 MB
    unsigned short* QK  = W16 + 4 * (size_t)CDIM * CDIM;            // 100.7 MB

    // 1) fp32 -> bf16
    cvt_feature_kernel<<<(MTOT * CDIM) / 4 / 256, 256, 0, stream>>>(feat, X16);
    cvt_weights_kernel<<<dim3((CDIM * CDIM) / 4 / 256, 4), 256, 0, stream>>>(
        Wq, Wk, Wqc, Wkc, W16);

    // 2) projections + RoPE (z order matches W16: Q, K, Q_cls, K_cls)
    proj_rope_kernel<<<dim3(CDIM / 128, MTOT / 128, 4), 256, 0, stream>>>(
        X16, W16, bq, bk, bqc, bkc, QK);

    // 3) logits, mask, scale
    logits_kernel<<<dim3(NTOK / 256, NTOK / 128, 16), 256, 0, stream>>>(
        QK, (float*)d_out);
}
